// ClassAttention_45234595561628
// MI455X (gfx1250) — matrix operand order repaired and checked by reading, not compiled, not run on hardware
//
#include <hip/hip_runtime.h>
#include <hip/hip_bf16.h>
#include <math.h>

// ---------------------------------------------------------------------------
// MHA: B=4, N=4096, C=384, H=6, D=64.  ~122 GFLOP, ~115MB traffic -> compute
// bound on MI455X; all GEMM stages on v_wmma_f32_16x16x32_bf16, flash-attn
// K/V tiles staged with gfx1250 async LDS DMA (ASYNCcnt) + double buffering.
// ---------------------------------------------------------------------------

#define BATCH 4
#define NTOK  4096
#define CDIM  384
#define HEADS 6
#define DHEAD 64
#define BH    (BATCH * HEADS)
#define MROWS (BATCH * NTOK)          // 16384
#define QKVC  (3 * CDIM)              // 1152

typedef __bf16 bf16;
typedef __attribute__((ext_vector_type(8)))  float v8f;
typedef __attribute__((ext_vector_type(16))) bf16  v16bf;
typedef __attribute__((ext_vector_type(8)))  bf16  v8bf;

// ---- fragment loaders (match CDNA5 16-bit WMMA VGPR layouts) --------------
// A 16x32: lane = row (lane&15); elems 0..7 -> K = (lane>>4)*8 + 0..7,
//          elems 8..15 -> K = 16 + (lane>>4)*8 + 0..7   (rowp points at K=0)
__device__ __forceinline__ v16bf ld_a_frag(const bf16* rowp, int hl) {
  v8bf lo = *(const v8bf*)(rowp + hl * 8);
  v8bf hi = *(const v8bf*)(rowp + 16 + hl * 8);
  return __builtin_shufflevector(lo, hi, 0,1,2,3,4,5,6,7,8,9,10,11,12,13,14,15);
}
// B 32x16: lane = col (lane&15); elems 0..7 -> K = (lane>>4)*8 + 0..7,
//          elems 8..15 -> K = 16 + (lane>>4)*8 + 0..7,
// read from B^T stored row-major (rowp = row `col` of B^T at K=0)
__device__ __forceinline__ v16bf ld_b_frag(const bf16* rowp, int hl) {
  v8bf lo = *(const v8bf*)(rowp + hl * 8);
  v8bf hi = *(const v8bf*)(rowp + 16 + hl * 8);
  return __builtin_shufflevector(lo, hi, 0,1,2,3,4,5,6,7,8,9,10,11,12,13,14,15);
}
__device__ __forceinline__ v8f wmma_bf16(v16bf a, v16bf b, v8f c) {
  return __builtin_amdgcn_wmma_f32_16x16x32_bf16(false, a, false, b, (short)0, c,
                                                 false, false);
}

// ---- gfx1250 async global->LDS DMA (tracked by ASYNCcnt) ------------------
// GV form: global_load_async_to_lds_b128  vdst_lds, v[addr_lo:addr_hi], off
// vdst carries the wave-relative LDS byte offset (low 32 bits of generic ptr).
__device__ __forceinline__ void async_ld_b128(const bf16* g, const bf16* l) {
  unsigned lds = (unsigned)(unsigned long long)(const void*)l;
  unsigned long long ga = (unsigned long long)(const void*)g;
  asm volatile("global_load_async_to_lds_b128 %0, %1, off"
               :: "v"(lds), "v"(ga) : "memory");
}
__device__ __forceinline__ void wait_async() {
  asm volatile("s_wait_asynccnt 0x0" ::: "memory");
}

// ---------------------------------------------------------------------------
// Cast / transpose prep kernels
// ---------------------------------------------------------------------------
__global__ void cast_f32_bf16_kernel(const float* __restrict__ src,
                                     bf16* __restrict__ dst, int n) {
  int stride = gridDim.x * blockDim.x;
  for (int i = blockIdx.x * blockDim.x + threadIdx.x; i < n; i += stride)
    dst[i] = (bf16)src[i];
}

// src[rows][cols] -> dst[cols][rows]  (store W^T so B-fragments are contiguous)
__global__ void transpose_cast_kernel(const float* __restrict__ src,
                                      bf16* __restrict__ dst, int rows, int cols) {
  int stride = gridDim.x * blockDim.x;
  int n = rows * cols;
  for (int i = blockIdx.x * blockDim.x + threadIdx.x; i < n; i += stride) {
    int r = i / cols, c = i % cols;
    dst[(size_t)c * rows + r] = (bf16)src[i];
  }
}

// ---------------------------------------------------------------------------
// QKV GEMM: qkv = x_bf16 [16384 x 384] @ Wqkv^T-fragments [1152 x 384]
// Each wave: 16 rows x 64 cols (one (s,h) chunk).  Q,K stored [BH][N][64],
// V stored transposed [BH][64][N] for the PV stage.
// ---------------------------------------------------------------------------
__global__ __launch_bounds__(256) void qkv_gemm_kernel(
    const bf16* __restrict__ X, const bf16* __restrict__ Wt,
    bf16* __restrict__ Qb, bf16* __restrict__ Kb, bf16* __restrict__ Vtb) {
  const int lane = threadIdx.x & 31;
  const int r = lane & 15, hl = lane >> 4;
  const int w = blockIdx.x * 8 + (threadIdx.x >> 5);   // 18432 waves total
  const int mtile = w / 18, grp = w % 18;
  const int s = grp / 6, h = grp % 6;
  const int mbase = mtile * 16;
  const int colbase = (s * 6 + h) * 64;                // s*384 + h*64

  const bf16* Ap = X + (size_t)(mbase + r) * CDIM;
  v8f acc[4];
#pragma unroll
  for (int nt = 0; nt < 4; ++nt) acc[nt] = (v8f)0.0f;

  for (int k = 0; k < CDIM; k += 32) {
    v16bf a = ld_a_frag(Ap + k, hl);
#pragma unroll
    for (int nt = 0; nt < 4; ++nt) {
      const bf16* Bp = Wt + (size_t)(colbase + nt * 16 + r) * CDIM + k;
      acc[nt] = wmma_bf16(a, ld_b_frag(Bp, hl), acc[nt]);
    }
  }

  const int b  = mbase / NTOK;
  const int n0 = mbase % NTOK;
  const int bh = b * HEADS + h;
  if (s < 2) {
    bf16* dst = (s == 0) ? Qb : Kb;
#pragma unroll
    for (int nt = 0; nt < 4; ++nt) {
      const int d = nt * 16 + r;
#pragma unroll
      for (int j = 0; j < 8; ++j) {
        const int n = n0 + hl * 8 + j;
        dst[((size_t)bh * NTOK + n) * DHEAD + d] = (bf16)acc[nt][j];
      }
    }
  } else {  // V transposed: Vt[bh][d][n] -> lane-consecutive n, packed store
#pragma unroll
    for (int nt = 0; nt < 4; ++nt) {
      const int d = nt * 16 + r;
      v8bf pk;
#pragma unroll
      for (int j = 0; j < 8; ++j) pk[j] = (bf16)acc[nt][j];
      *(v8bf*)(Vtb + ((size_t)bh * DHEAD + d) * NTOK + n0 + hl * 8) = pk;
    }
  }
}

// ---------------------------------------------------------------------------
// Flash attention.  Block = 4 waves, each wave owns 16 queries (64 per block).
// 32-key blocks; K and V^T tiles double-buffered in LDS via async DMA and
// shared by all 4 waves.  Scores S = Q K^T (C-layout: row=q slot, col=key in
// lane); online softmax with __shfl_xor butterflies across 16-lane halves.
// PV computed as out^T = V^T P^T (P round-trips a wave-private LDS tile) so
// the C-tile has 8 consecutive d per lane -> packed 16B stores.
// ---------------------------------------------------------------------------
__global__ __launch_bounds__(128) void flash_attn_kernel(
    const bf16* __restrict__ Q, const bf16* __restrict__ K,
    const bf16* __restrict__ Vt, bf16* __restrict__ AO) {
  __shared__ __align__(32) bf16  Ksh[2][32 * DHEAD];  // [buf][key][d]  8KB
  __shared__ __align__(32) bf16  Vsh[2][DHEAD * 32];  // [buf][d][key]  8KB
  __shared__ __align__(32) bf16  Psh[4][16 * 32];     // per-wave P [q][key]
  __shared__ float Fsh[4][16];                        // per-wave row stats

  const int tid  = threadIdx.x;
  const int wid  = tid >> 5, lane = tid & 31;
  const int r = lane & 15, hl = lane >> 4;
  const int bh    = blockIdx.x / (NTOK / 64);
  const int qblk  = blockIdx.x % (NTOK / 64);
  const int qbase = qblk * 64 + wid * 16;

  const float alpha = 0.125f * 1.44269504088896340736f;  // SCALE * log2(e)

  const bf16* Kg0  = K  + (size_t)bh * NTOK * DHEAD;
  const bf16* Vtg0 = Vt + (size_t)bh * DHEAD * NTOK;

  // async-stage one 32-key tile (K rows + V^T rows) into LDS buffer `buf`
  auto stage = [&](int kb, int buf) {
    const bf16* Kg = Kg0 + (size_t)kb * DHEAD;
    const int e = tid * 16;                     // 2048 elems / 128 threads
    async_ld_b128(Kg + e,     &Ksh[buf][e]);
    async_ld_b128(Kg + e + 8, &Ksh[buf][e + 8]);
    const int d = tid >> 1, part = tid & 1;
    const bf16* Vg = Vtg0 + (size_t)d * NTOK + kb + part * 16;
    const bf16* Vd = &Vsh[buf][d * 32 + part * 16];
    async_ld_b128(Vg,     Vd);
    async_ld_b128(Vg + 8, Vd + 8);
  };

  // Q fragments for this wave's 16 rows (resident for the whole key loop)
  const bf16* Qp = Q + ((size_t)bh * NTOK + qbase + r) * DHEAD;
  const v16bf aq0 = ld_a_frag(Qp + 0,  hl);
  const v16bf aq1 = ld_a_frag(Qp + 32, hl);

  float mrow[8], lrow[8];
#pragma unroll
  for (int j = 0; j < 8; ++j) { mrow[j] = -1.0e30f; lrow[j] = 0.0f; }
  v8f acc[4];
#pragma unroll
  for (int mt = 0; mt < 4; ++mt) acc[mt] = (v8f)0.0f;

  stage(0, 0);
  wait_async();
  __syncthreads();

  for (int kb = 0; kb < NTOK; kb += 32) {
    const int cur = (kb >> 5) & 1;
    if (kb + 32 < NTOK) stage(kb + 32, cur ^ 1);   // overlap next tile DMA

    // ---- scores: two 16x16 tiles over keys kb..kb+31 ----------------------
    const bf16* Kt = Ksh[cur];
    v16bf bk00 = ld_b_frag(Kt + (0 * 16 + r) * DHEAD + 0,  hl);
    v16bf bk01 = ld_b_frag(Kt + (0 * 16 + r) * DHEAD + 32, hl);
    v16bf bk10 = ld_b_frag(Kt + (1 * 16 + r) * DHEAD + 0,  hl);
    v16bf bk11 = ld_b_frag(Kt + (1 * 16 + r) * DHEAD + 32, hl);
    v8f s0 = wmma_bf16(aq0, bk00, (v8f)0.0f);
    s0     = wmma_bf16(aq1, bk01, s0);
    v8f s1 = wmma_bf16(aq0, bk10, (v8f)0.0f);
    s1     = wmma_bf16(aq1, bk11, s1);

    // ---- online softmax over this key block ------------------------------
#pragma unroll
    for (int j = 0; j < 8; ++j) {
      float a0 = s0[j] * alpha, a1 = s1[j] * alpha;
      float c = fmaxf(a0, a1);
      c = fmaxf(c, __shfl_xor(c, 1));
      c = fmaxf(c, __shfl_xor(c, 2));
      c = fmaxf(c, __shfl_xor(c, 4));
      c = fmaxf(c, __shfl_xor(c, 8));
      const float mn = fmaxf(mrow[j], c);
      const float f  = __builtin_exp2f(mrow[j] - mn);
      mrow[j] = mn;
      const float p0 = __builtin_exp2f(a0 - mn);
      const float p1 = __builtin_exp2f(a1 - mn);
      float rs = p0 + p1;
      rs += __shfl_xor(rs, 1);
      rs += __shfl_xor(rs, 2);
      rs += __shfl_xor(rs, 4);
      rs += __shfl_xor(rs, 8);
      lrow[j] = lrow[j] * f + rs;
      const int q = hl * 8 + j;                  // row index of this slot
      Psh[wid][q * 32 + r]      = (bf16)p0;      // P[q][key] row-major
      Psh[wid][q * 32 + 16 + r] = (bf16)p1;
      if (r == 0) Fsh[wid][q] = f;               // broadcast rescale factor
    }
    const float fq = Fsh[wid][r];                // factor for q = lane&15
#pragma unroll
    for (int mt = 0; mt < 4; ++mt)
#pragma unroll
      for (int j = 0; j < 8; ++j) acc[mt][j] *= fq;

    // ---- out^T += V^T-tile @ P^T  (K dim = 32 keys, one WMMA per d-tile) --
    const bf16* Vc = Vsh[cur];
#pragma unroll
    for (int mt = 0; mt < 4; ++mt) {
      const int d = mt * 16 + r;
      v8bf vlo = *(const v8bf*)(Vc + d * 32 + hl * 8);
      v8bf vhi = *(const v8bf*)(Vc + d * 32 + 16 + hl * 8);
      v16bf av = __builtin_shufflevector(vlo, vhi,
                   0,1,2,3,4,5,6,7,8,9,10,11,12,13,14,15);
      v16bf bp = ld_b_frag(&Psh[wid][r * 32], hl);
      acc[mt] = wmma_bf16(av, bp, acc[mt]);
    }
    wait_async();            // next tile's DMA landed in LDS
    __syncthreads();
  }

  // ---- finalize: 1/l per query, packed bf16 store to [B,N,H*D] ------------
#pragma unroll
  for (int j = 0; j < 8; ++j)
    if (r == 0) Fsh[wid][hl * 8 + j] = lrow[j];
  const float linv = 1.0f / Fsh[wid][r];

  const int b = bh / HEADS, h = bh % HEADS;
  bf16* outp = AO + ((size_t)(b * NTOK + qbase + r)) * CDIM + h * DHEAD;
#pragma unroll
  for (int mt = 0; mt < 4; ++mt) {
    v8bf pk;
#pragma unroll
    for (int j = 0; j < 8; ++j) pk[j] = (bf16)(acc[mt][j] * linv);
    *(v8bf*)(outp + mt * 16 + hl * 8) = pk;      // 8 consecutive d values
  }
}

// ---------------------------------------------------------------------------
// Output projection: out = AO [16384 x 384] @ Wproj + bias  (f32 out)
// ---------------------------------------------------------------------------
__global__ __launch_bounds__(256) void proj_gemm_kernel(
    const bf16* __restrict__ AO, const bf16* __restrict__ Wt,
    const float* __restrict__ bias, float* __restrict__ out) {
  const int lane = threadIdx.x & 31;
  const int r = lane & 15, hl = lane >> 4;
  const int w = blockIdx.x * 8 + (threadIdx.x >> 5);   // 6144 waves
  const int mtile = w / 6, grp = w % 6;
  const int mbase = mtile * 16, nbase = grp * 64;

  const bf16* Ap = AO + (size_t)(mbase + r) * CDIM;
  v8f acc[4];
#pragma unroll
  for (int nt = 0; nt < 4; ++nt) acc[nt] = (v8f)0.0f;

  for (int k = 0; k < CDIM; k += 32) {
    v16bf a = ld_a_frag(Ap + k, hl);
#pragma unroll
    for (int nt = 0; nt < 4; ++nt) {
      const bf16* Bp = Wt + (size_t)(nbase + nt * 16 + r) * CDIM + k;
      acc[nt] = wmma_bf16(a, ld_b_frag(Bp, hl), acc[nt]);
    }
  }
#pragma unroll
  for (int nt = 0; nt < 4; ++nt) {
    const int col = nbase + nt * 16 + r;
    const float bv = bias[col];
#pragma unroll
    for (int j = 0; j < 8; ++j)
      out[(size_t)(mbase + hl * 8 + j) * CDIM + col] = acc[nt][j] + bv;
  }
}

// ---------------------------------------------------------------------------
extern "C" void kernel_launch(void* const* d_in, const int* in_sizes, int n_in,
                              void* d_out, int out_size, void* d_ws, size_t ws_size,
                              hipStream_t stream) {
  const float* x      = (const float*)d_in[0];
  const float* w_qkv  = (const float*)d_in[1];
  const float* w_proj = (const float*)d_in[2];
  const float* b_proj = (const float*)d_in[3];
  float* out = (float*)d_out;

  char* wsp = (char*)d_ws;
  size_t off = 0;
  auto take = [&](size_t bytes) -> char* {
    char* p = wsp + off;
    off += (bytes + 255) & ~(size_t)255;
    return p;
  };
  bf16* xb     = (bf16*)take((size_t)MROWS * CDIM * 2);
  bf16* wqkvT  = (bf16*)take((size_t)QKVC * CDIM * 2);
  bf16* wprojT = (bf16*)take((size_t)CDIM * CDIM * 2);
  bf16* Qb     = (bf16*)take((size_t)BH * NTOK * DHEAD * 2);
  bf16* Kb     = (bf16*)take((size_t)BH * NTOK * DHEAD * 2);
  bf16* Vtb    = (bf16*)take((size_t)BH * DHEAD * NTOK * 2);
  bf16* AO     = (bf16*)take((size_t)MROWS * CDIM * 2);
  (void)in_sizes; (void)n_in; (void)out_size; (void)ws_size;

  cast_f32_bf16_kernel<<<2048, 256, 0, stream>>>(x, xb, MROWS * CDIM);
  transpose_cast_kernel<<<864, 256, 0, stream>>>(w_qkv, wqkvT, CDIM, QKVC);
  transpose_cast_kernel<<<288, 256, 0, stream>>>(w_proj, wprojT, CDIM, CDIM);

  qkv_gemm_kernel<<<2304, 256, 0, stream>>>(xb, wqkvT, Qb, Kb, Vtb);
  flash_attn_kernel<<<BH * (NTOK / 64), 128, 0, stream>>>(Qb, Kb, Vtb, AO);
  proj_gemm_kernel<<<768, 256, 0, stream>>>(AO, wprojT, b_proj, out);
}
